// FixedWindowModelLSTM_28037546508993
// MI455X (gfx1250) — compile-verified, matrix operand order checked by reading, not run on hardware
//
#include <hip/hip_runtime.h>
#include <hip/hip_bf16.h>

typedef __attribute__((ext_vector_type(16))) _Float16     v16h;
typedef __attribute__((ext_vector_type(8)))  _Float16     v8h;
typedef __attribute__((ext_vector_type(8)))  float        v8f;
typedef __attribute__((ext_vector_type(4)))  unsigned int v4u;
typedef __attribute__((ext_vector_type(8)))  int          v8i;
typedef __attribute__((ext_vector_type(4)))  int          v4i;

// Problem constants
#define NTOK   262144
#define DW     100
#define DT     25
#define KDIM   125
#define KPAD   128            // K padded to multiple of 32
#define HID    100
#define PC_H   304            // packed cols per direction: 3 gates * 100, padded to 19 tiles
#define NCT_H  19             // column tiles per direction
#define KT     4              // K tiles of 32
#define TM     64             // tokens per block
#define GSTRIDE 312           // 304 + 8 pad (bank-conflict break)
#define HALF_BYTES  (NCT_H * KT * 32 * 32)   // 77,824 B per direction of packed B
#define TILE_ELEMS  (HALF_BYTES / 8)         // 9,728 8-byte TDM elements (fits 16-bit tile_dim0)
#define LOG2E  1.4426950408889634f

// torch gate order i,f,g,o -> we keep i(row 0), g(row 200), o(row 300); f is dead (c0=0)
__device__ __forceinline__ int gate_src_row(int r /*0..299*/) {
    int gate = r / HID, hh = r % HID;
    int off = (gate == 0) ? 0 : ((gate == 1) ? 200 : 300);
    return off + hh;
}

// fast transcendental helpers: single v_exp_f32 / v_rcp_f32 TRANS ops
__device__ __forceinline__ float fast_sigmoid(float x) {
    return __builtin_amdgcn_rcpf(1.0f + __builtin_amdgcn_exp2f(-x * LOG2E));
}
__device__ __forceinline__ float fast_tanh(float x) {
    // tanh(x) = 1 - 2/(exp(2x)+1); exp2 overflow -> rcp(inf)=0 -> +1, underflow -> -1
    return 1.0f - 2.0f * __builtin_amdgcn_rcpf(__builtin_amdgcn_exp2f(x * (2.0f * LOG2E)) + 1.0f);
}

// ---------------------------------------------------------------------------
// Pack W_ih_{f,b} (f32 [400,125]) -> f16 B fragments, one contiguous slab per
// direction: Bpack[((((h*19 + j)*4 + kb)*32 + lane) << 4) + half].
// Lane mapping for v_wmma_f32_16x16x32_f16 B (32x16, K-major per lane):
//   lanes 0-15  : N = lane,    halves 0-7 = K 0-7,   halves 8-15 = K 16-23
//   lanes 16-31 : N = lane-16, halves 0-7 = K 8-15,  halves 8-15 = K 24-31
// ---------------------------------------------------------------------------
__global__ void pack_weights_kernel(const float* __restrict__ Wf,
                                    const float* __restrict__ Wb,
                                    _Float16* __restrict__ Bpack) {
    int idx = blockIdx.x * blockDim.x + threadIdx.x;   // 2 * PC_H * KPAD
    if (idx >= 2 * PC_H * KPAD) return;
    int p = idx >> 7;          // packed column 0..607
    int k = idx & (KPAD - 1);  // 0..127
    int h = p / PC_H;          // direction
    int r = p % PC_H;          // col within direction (0..303)
    float v = 0.0f;
    if (r < 3 * HID && k < KDIM) {
        int srow = gate_src_row(r);
        v = (h ? Wb : Wf)[srow * KDIM + k];
    }
    int j  = r >> 4;           // col tile within direction
    int l  = r & 15;
    int kb = k >> 5;
    int ko = k & 31;
    int lane = l + ((((ko >> 3) & 1) != 0) ? 16 : 0);
    int half = (ko & 7) + ((ko >= 16) ? 8 : 0);
    Bpack[((((h * NCT_H + j) * KT + kb) * 32 + lane) << 4) + half] = (_Float16)v;
}

__global__ void pack_bias_kernel(const float* __restrict__ bif, const float* __restrict__ bhf,
                                 const float* __restrict__ bib, const float* __restrict__ bhb,
                                 float* __restrict__ bias) {
    int p = blockIdx.x * blockDim.x + threadIdx.x;     // 2 * PC_H
    if (p >= 2 * PC_H) return;
    int h = p / PC_H, r = p % PC_H;
    float v = 0.0f;
    if (r < 3 * HID) {
        int srow = gate_src_row(r);
        v = h ? (bib[srow] + bhb[srow]) : (bif[srow] + bhf[srow]);
    }
    bias[p] = v;
}

// ---------------------------------------------------------------------------
// Fused: gather-embed -> f16 LDS A-panel -> TDM-stage B half into LDS ->
// WMMA gate GEMM -> LDS gates -> sigmoid/tanh epilogue -> f32 output.
// One block = 64 tokens, 8 wave32s. LDS: 16K (A) + 76K (B) + 78K (G) = 174KB.
// ---------------------------------------------------------------------------
__global__ __launch_bounds__(256, 1)
void lstm_fused_wmma_kernel(const int*   __restrict__ sentence,
                            const int*   __restrict__ tags,
                            const float* __restrict__ Ew,
                            const float* __restrict__ Et,
                            const _Float16* __restrict__ Bpack,
                            const float* __restrict__ bias,
                            float* __restrict__ out) {
    extern __shared__ char smem[];
    _Float16* sA = (_Float16*)smem;                            // [TM][KPAD] f16   = 16,384 B
    _Float16* sB = (_Float16*)(smem + TM * KPAD * 2);          // one B half      = 77,824 B
    float*    sG = (float*)(smem + TM * KPAD * 2 + HALF_BYTES);// [TM][GSTRIDE]   = 79,872 B

    const int tid  = threadIdx.x;
    const int lane = tid & 31;
    const int wave = tid >> 5;
    const int tokBase = blockIdx.x * TM;

    // ---- Phase 1: gather embeddings, convert f32 -> f16, K padded to 128 ----
    {
        int tt = tid >> 2;            // token 0..63 (4 threads per token)
        int q  = tid & 3;             // quarter of K: [q*32, q*32+32)
        int tok = tokBase + tt;
        const float* ewr = Ew + (long)sentence[tok] * DW;   // row is 400B, 16B-aligned
        const float* etr = Et + (long)tags[tok] * DT;
        if (q < 3) {                  // K 0..95: fully inside E_w row, vector loads
            const float4* src = (const float4*)(ewr + q * 32);
            #pragma unroll
            for (int kk = 0; kk < 8; ++kk) {
                float4 v = src[kk];
                int k = q * 32 + kk * 4;
                sA[tt * KPAD + k + 0] = (_Float16)v.x;
                sA[tt * KPAD + k + 1] = (_Float16)v.y;
                sA[tt * KPAD + k + 2] = (_Float16)v.z;
                sA[tt * KPAD + k + 3] = (_Float16)v.w;
            }
        } else {                      // K 96..127: E_w tail, E_t row, zero pad
            #pragma unroll
            for (int kk = 0; kk < 32; ++kk) {
                int k = 96 + kk;
                float v;
                if      (k < DW)   v = ewr[k];
                else if (k < KDIM) v = etr[k - DW];
                else               v = 0.0f;
                sA[tt * KPAD + k] = (_Float16)v;
            }
        }
    }

    const int arow = lane & 15;
    const int asel = (lane >> 4) * 8;

    for (int h = 0; h < 2; ++h) {
        // ---- Phase 2a: TDM async-stage this direction's B slab into LDS ----
        if (wave == 0) {
            unsigned long long ga =
                (unsigned long long)(uintptr_t)Bpack + (unsigned long long)h * HALF_BYTES;
            unsigned int ldsb = (unsigned int)(uintptr_t)sB;
            v4u g0;
            g0[0] = 1u;                                            // count=1 (valid), user mode
            g0[1] = ldsb;                                          // lds_addr [63:32]
            g0[2] = (unsigned int)ga;                              // global_addr [95:64]
            g0[3] = (unsigned int)((ga >> 32) & 0x01FFFFFFull)     // global_addr [120:96]
                  | (2u << 30);                                    // type=2 ("image") [127:126]
            v8i g1;
            g1[0] = (int)(3u << 16);                               // data_size=8B; wg_mask=0
            g1[1] = (int)((TILE_ELEMS & 0xFFFF) << 16);            // tensor_dim0[15:0] @63:48
            g1[2] = (int)((TILE_ELEMS >> 16) & 0xFFFF) | (1 << 16);// dim0 hi | tensor_dim1=1
            g1[3] = (int)((TILE_ELEMS & 0xFFFF) << 16);            // tile_dim0 @127:112
            g1[4] = 0;                                             // tile_dim1=0, tile_dim2=0
            g1[5] = TILE_ELEMS;                                    // tensor_dim0_stride lo32
            g1[6] = 0;
            g1[7] = 0;
            v4i z4 = {0, 0, 0, 0};
            v8i z8 = {0, 0, 0, 0, 0, 0, 0, 0};
            __builtin_amdgcn_tensor_load_to_lds(g0, g1, z4, z4, z8, 0);
            __builtin_amdgcn_s_wait_tensorcnt(0);
        }
        __syncthreads();   // publish sA (h==0) and sB to all waves

        // ---- Phase 2b: gates[64 x 304] = A[64 x 128] * B[128 x 304] via WMMA ----
        for (int t = wave; t < 4 * NCT_H; t += 8) {
            int m = t / NCT_H;            // row tile
            int j = t % NCT_H;            // col tile
            v8f c = {};
            #pragma unroll
            for (int kb = 0; kb < KT; ++kb) {
                // A fragment from LDS
                const _Float16* ap = sA + (m * 16 + arow) * KPAD + kb * 32 + asel;
                v8h alo = *(const v8h*)(ap);
                v8h ahi = *(const v8h*)(ap + 16);
                v16h a;
                #pragma unroll
                for (int i = 0; i < 8; ++i) { a[i] = alo[i]; a[i + 8] = ahi[i]; }
                // B fragment from TDM-staged LDS (contiguous 32B per lane)
                v16h b = *(const v16h*)(sB + (((j * KT + kb) * 32 + lane) << 4));
                c = __builtin_amdgcn_wmma_f32_16x16x32_f16(
                        /*neg_a=*/false, a, /*neg_b=*/false, b,
                        /*c_mod=*/(short)0, c, /*reuse_a=*/false, /*reuse_b=*/false);
            }
            // C layout: VGPR r -> M = r (lanes 0-15) / r+8 (lanes 16-31), N = lane&15
            const int crow = m * 16 + ((lane >= 16) ? 8 : 0);
            const int ccol = j * 16 + (lane & 15);
            #pragma unroll
            for (int r = 0; r < 8; ++r)
                sG[(crow + r) * GSTRIDE + ccol] = c[r];
        }
        __syncthreads();

        // ---- Phase 3: epilogue h = sigmoid(o) * tanh(sigmoid(i) * tanh(g)) ----
        // fixed mapping: 64 tokens x 4 col-quarters of 25 == 256 threads (no div/mod)
        {
            const float* bh = bias + h * PC_H;
            const int t  = tid >> 2;
            const int c0 = (tid & 3) * 25;
            const float* g = sG + t * GSTRIDE;
            float* o = out + (long)(tokBase + t) * (2 * HID) + h * HID;
            #pragma unroll 5
            for (int i = 0; i < 25; ++i) {
                int hh = c0 + i;
                float gi = g[      hh] + bh[      hh];
                float gg = g[100 + hh] + bh[100 + hh];
                float go = g[200 + hh] + bh[200 + hh];
                float cc = fast_sigmoid(gi) * fast_tanh(gg);
                o[hh] = fast_sigmoid(go) * fast_tanh(cc);
            }
        }
        __syncthreads();   // protect sB/sG before the next half overwrites them
    }
}

extern "C" void kernel_launch(void* const* d_in, const int* in_sizes, int n_in,
                              void* d_out, int out_size, void* d_ws, size_t ws_size,
                              hipStream_t stream) {
    const int*   sentence = (const int*)  d_in[0];
    const int*   tags     = (const int*)  d_in[1];
    const float* Ew       = (const float*)d_in[2];
    const float* Et       = (const float*)d_in[3];
    const float* Wf       = (const float*)d_in[4];
    const float* bif      = (const float*)d_in[5];
    const float* bhf      = (const float*)d_in[6];
    const float* Wb       = (const float*)d_in[7];
    const float* bib      = (const float*)d_in[8];
    const float* bhb      = (const float*)d_in[9];
    float* out = (float*)d_out;

    // workspace partition
    _Float16* Bpack = (_Float16*)d_ws;                         // 2 * 77,824 B
    float*    bias  = (float*)((char*)d_ws + 2 * HALF_BYTES);  // 608 * 4 B

    pack_weights_kernel<<<(2 * PC_H * KPAD + 255) / 256, 256, 0, stream>>>(Wf, Wb, Bpack);
    pack_bias_kernel<<<(2 * PC_H + 255) / 256, 256, 0, stream>>>(bif, bhf, bib, bhb, bias);

    size_t shmem = (size_t)TM * KPAD * 2      // sA  16,384
                 + (size_t)HALF_BYTES         // sB  77,824
                 + (size_t)TM * GSTRIDE * 4;  // sG  79,872  => 174,080 B of 320KB WGP LDS
    lstm_fused_wmma_kernel<<<NTOK / TM, 256, shmem, stream>>>(
        sentence, tags, Ew, Et, Bpack, bias, out);
}
